// MultiHeadAttention_15298673509029
// MI455X (gfx1250) — matrix operand order repaired and checked by reading, not compiled, not run on hardware
//
#include <hip/hip_runtime.h>

typedef __attribute__((ext_vector_type(16))) _Float16 v16h;
typedef __attribute__((ext_vector_type(8)))  _Float16 v8h;
typedef __attribute__((ext_vector_type(8)))  float    v8f;

#define S_LEN 1024
#define DKQ   8
#define DV    8

__global__ __launch_bounds__(256)
void mha_flash_wmma_gfx1250(const float* __restrict__ x,    // [4,1024,4]
                            const float* __restrict__ Wq,   // [32,4,8]
                            const float* __restrict__ Wk,   // [32,4,8]
                            const float* __restrict__ Wv,   // [32,4,8]
                            float* __restrict__ out)        // [4,1024,256]
{
    // LDS: per-(b,h) projected tensors in f16.
    __shared__ __align__(16) _Float16 Qlds[S_LEN * DKQ];   // [s][d] row-major
    __shared__ __align__(16) _Float16 Klds[S_LEN * DKQ];   // [s][d] row-major
    __shared__ __align__(16) _Float16 vT  [DV * S_LEN];    // [d][s] (transposed V)

    const int b    = blockIdx.x >> 5;   // H = 32
    const int h    = blockIdx.x & 31;
    const int tid  = threadIdx.x;
    const int lane = tid & 31;
    const int wave = tid >> 5;

    // ---------------- Phase 1: cooperative QKV projection (d_in = 4) --------
    const float* wqh = Wq + h * 32;     // [i][d] = wqh[i*8 + d]
    const float* wkh = Wk + h * 32;
    const float* wvh = Wv + h * 32;

    for (int idx = tid; idx < S_LEN * DKQ; idx += 256) {
        const int s = idx >> 3;
        const int d = idx & 7;
        const float4 xv = *(const float4*)(x + (b * S_LEN + s) * 4);
        const float qf = xv.x * wqh[d] + xv.y * wqh[8 + d] + xv.z * wqh[16 + d] + xv.w * wqh[24 + d];
        const float kf = xv.x * wkh[d] + xv.y * wkh[8 + d] + xv.z * wkh[16 + d] + xv.w * wkh[24 + d];
        const float vf = xv.x * wvh[d] + xv.y * wvh[8 + d] + xv.z * wvh[16 + d] + xv.w * wvh[24 + d];
        Qlds[idx]         = (_Float16)qf;
        Klds[idx]         = (_Float16)kf;
        vT[d * S_LEN + s] = (_Float16)vf;
    }
    __syncthreads();

    // ---------------- Phase 2: flash attention, one wave per 16-query tile,
    // ---------------- 32 keys per online-softmax iteration ------------------
    const float scale = 0.35355339059327373f;   // 1/sqrt(8)
    const int   hi8   = (lane >= 16) ? 8 : 0;

    for (int qt = wave; qt < S_LEN / 16; qt += 8) {
        // B operand of S^T = K * Q^T : B[k][n] = q[n][k].
        // B as loaded here: lane n (<16) holds K=0..7 (q[n][0..7]) in halves 0..7;
        // halves 8..15 and lanes 16..31 are zero (d_kq = 8).
        v16h bq = {};
        if (lane < 16) {
            const v8h r = *(const v8h*)&Qlds[(qt * 16 + lane) * DKQ];
            #pragma unroll
            for (int i = 0; i < 8; ++i) bq[i] = r[i];
        }

        float mrun = -__builtin_inff();
        float lrun = 0.0f;
        v8f   o    = {};

        for (int j = 0; j < S_LEN / 32; ++j) {
            const int t0 = j * 32;

            // A operands: two K-tiles (M = key index, K-dim = d_kq zero-padded).
            v16h ak0 = {}, ak1 = {};
            if (lane < 16) {
                const v8h r0 = *(const v8h*)&Klds[(t0 + lane) * DKQ];
                const v8h r1 = *(const v8h*)&Klds[(t0 + 16 + lane) * DKQ];
                #pragma unroll
                for (int i = 0; i < 8; ++i) { ak0[i] = r0[i]; ak1[i] = r1[i]; }
            }

            v8f cz = {};
            // sc0[r] @ lane l = S^T[t0    + r + 8*(l>=16)][m = l%16]
            // sc1[r] @ lane l = S^T[t0+16 + r + 8*(l>=16)][m = l%16]
            v8f sc0 = __builtin_amdgcn_wmma_f32_16x16x32_f16(
                false, ak0, false, bq, (short)0, cz, false, false);
            v8f sc1 = __builtin_amdgcn_wmma_f32_16x16x32_f16(
                false, ak1, false, bq, (short)0, cz, false, false);

            #pragma unroll
            for (int r = 0; r < 8; ++r) { sc0[r] *= scale; sc1[r] *= scale; }

            // online softmax over 32 keys: in-lane over 16 values + cross-half
            float tmax = fmaxf(sc0[0], sc1[0]);
            #pragma unroll
            for (int r = 1; r < 8; ++r)
                tmax = fmaxf(tmax, fmaxf(sc0[r], sc1[r]));
            tmax = fmaxf(tmax, __shfl_xor(tmax, 16, 32));

            const float mnew  = fmaxf(mrun, tmax);
            const float alpha = __expf(mrun - mnew);   // first iter: exp(-inf)=0

            float tsum = 0.0f;
            v16h  ap;
            // Pᵀ in C-layout IS the A-layout of P for K=32:
            //   halves 0..7  = keys t0+{0..7 | 8..15}   (per lane half) = p0
            //   halves 8..15 = keys t0+{16..23 | 24..31}(per lane half) = p1
            #pragma unroll
            for (int r = 0; r < 8; ++r) {
                const float p0 = __expf(sc0[r] - mnew);
                const float p1 = __expf(sc1[r] - mnew);
                tsum += p0 + p1;
                ap[r]     = (_Float16)p0;
                ap[8 + r] = (_Float16)p1;
            }
            tsum += __shfl_xor(tsum, 16, 32);
            lrun  = lrun * alpha + tsum;
            mrun  = mnew;

            // rescale o (query index = r + 8*half in o's layout): broadcast alpha
            #pragma unroll
            for (int r = 0; r < 8; ++r) {
                const float ar = __shfl(alpha, r + hi8, 32);
                o[r] *= ar;
            }

            // B operand: V tile with K=32: B[k][n] = vT[n][t0+k].
            //   lanes 0..15  hold K=0..7,16..23  -> vT[n][t0+{0..7, 16..23}]
            //   lanes 16..31 hold K=8..15,24..31 -> vT[n][t0+{8..15, 24..31}]
            v16h bv = {};
            {
                const int n = lane & 15;
                if (n < 8) {
                    const v8h* p = (const v8h*)&vT[n * S_LEN + t0 + hi8];
                    const v8h lo = p[0];
                    const v8h hi = p[2];
                    #pragma unroll
                    for (int i = 0; i < 8; ++i) { bv[i] = lo[i]; bv[8 + i] = hi[i]; }
                }
            }

            o = __builtin_amdgcn_wmma_f32_16x16x32_f16(
                false, ap, false, bv, (short)0, o, false, false);
        }

        // epilogue: divide by running sum, write [b, s, h*8 + n]
        #pragma unroll
        for (int r = 0; r < 8; ++r) {
            const float lr = __shfl(lrun, r + hi8, 32);
            o[r] /= lr;
        }
        const int n = lane & 15;
        if (n < 8) {
            #pragma unroll
            for (int r = 0; r < 8; ++r) {
                const int sq = qt * 16 + r + hi8;
                out[(b * S_LEN + sq) * 256 + h * 8 + n] = o[r];
            }
        }
    }
}

extern "C" void kernel_launch(void* const* d_in, const int* in_sizes, int n_in,
                              void* d_out, int out_size, void* d_ws, size_t ws_size,
                              hipStream_t stream) {
    (void)in_sizes; (void)n_in; (void)out_size; (void)d_ws; (void)ws_size;
    const float* x  = (const float*)d_in[0];
    const float* Wq = (const float*)d_in[1];
    const float* Wk = (const float*)d_in[2];
    const float* Wv = (const float*)d_in[3];
    float* out = (float*)d_out;
    hipLaunchKernelGGL(mha_flash_wmma_gfx1250, dim3(4 * 32), dim3(256), 0, stream,
                       x, Wq, Wk, Wv, out);
}
